// GCNemb_69561290326351
// MI455X (gfx1250) — matrix operand order repaired and checked by reading, not compiled, not run on hardware
//
#include <hip/hip_runtime.h>
#include <hip/hip_bf16.h>
#include <stdint.h>

// ---- CDNA5 WMMA vector types (wave32) ----
typedef __attribute__((ext_vector_type(16))) __bf16 v16bf;
typedef __attribute__((ext_vector_type(8)))  __bf16 v8bf;
typedef __attribute__((ext_vector_type(8)))  float  v8f;

#define NN 100000
#define NE 3200000LL

// ---------------- degree / normalization ----------------
__global__ void k_init_ones(float* deg, int n) {
    int i = blockIdx.x * blockDim.x + threadIdx.x;
    if (i < n) deg[i] = 1.0f;               // self-loop contributes 1
}

__global__ void k_deg_count(const long long* __restrict__ col, float* deg, long long e) {
    long long t = (long long)blockIdx.x * blockDim.x + threadIdx.x;
    if (t < e) atomicAdd(&deg[col[t]], 1.0f);
}

__global__ void k_dinv(float* deg, int n) {
    int i = blockIdx.x * blockDim.x + threadIdx.x;
    if (i < n) {
        float d = deg[i];
        deg[i] = (d > 0.0f) ? rsqrtf(d) : 0.0f;   // deg >= 1 always (self-loop)
    }
}

// ---------------- fp32 -> bf16 staging for layer-0 input (K=3 padded to 32) -------------
__global__ void k_conv_a_bf16(const float* __restrict__ src, __bf16* __restrict__ dst,
                              int n, int K, int Kpad) {
    long long t = (long long)blockIdx.x * blockDim.x + threadIdx.x;
    long long total = (long long)n * Kpad;
    if (t >= total) return;
    int k = (int)(t % Kpad);
    long long r = t / Kpad;
    float v = (k < K) ? src[r * K + k] : 0.0f;
    dst[t] = (__bf16)v;
}

// W is (K x OUT) row-major fp32; produce Wt (OUTpad x Kpad) row-major bf16, zero-padded
__global__ void k_conv_wt_bf16(const float* __restrict__ Wsrc, __bf16* __restrict__ dst,
                               int K, int OUT, int Kpad, int OUTpad) {
    long long t = (long long)blockIdx.x * blockDim.x + threadIdx.x;
    long long total = (long long)OUTpad * Kpad;
    if (t >= total) return;
    int k = (int)(t % Kpad);
    int n = (int)(t / Kpad);
    float v = (k < K && n < OUT) ? Wsrc[(size_t)k * OUT + n] : 0.0f;
    dst[t] = (__bf16)v;
}

// ---------------- WMMA GEMM, 16x64 per wave, fused self-loop epilogue ----------------
// H[N x OUT] = A[N x Kpad] * Wt^T ; AGG = H * dinv(row)^2   (agg init, no pre-zero pass)
__global__ void __launch_bounds__(256) k_gemm_wmma_bf16(
        const __bf16* __restrict__ A,    // N x Kpad, row-major bf16
        const __bf16* __restrict__ Wt,   // OUTpad64 x Kpad, row-major bf16 (W transposed)
        const float* __restrict__ dinv,
        float* __restrict__ H,           // N x OUT, fp32
        float* __restrict__ AGG,         // N x OUT, fp32
        int n, int Kpad, int OUT, int tilesN4) {
    const int wave = threadIdx.x >> 5;
    const int lane = threadIdx.x & 31;
    const int tilesM = n >> 4;                        // N % 16 == 0 (100000 = 6250*16)
    const int tile = blockIdx.x * (blockDim.x >> 5) + wave;
    if (tile >= tilesM * tilesN4) return;             // wave-uniform: EXEC stays all-1s
    const int tm = tile / tilesN4;
    const int tn = tile % tilesN4;
    const int m0 = tm << 4;
    const int n0 = tn << 6;                           // 64 output cols per wave
    const int half = lane >> 4;                       // 0: lanes 0-15, 1: lanes 16-31
    const int l16  = lane & 15;

    const __bf16* arow = A  + (size_t)(m0 + l16) * Kpad;
    const __bf16* bbas = Wt + (size_t)(n0 + l16) * Kpad;
    const size_t  bstp = (size_t)16 * Kpad;           // 16 B-rows between sub-tiles

    v8f acc[4] = {{}, {}, {}, {}};
    for (int k0 = 0; k0 < Kpad; k0 += 32) {
        // A 16x32 fragment: elems 0-7 -> K = k0 + half*16 + i ; elems 8-15 -> K = k0+half*16+8+i
        v8bf alo = *(const v8bf*)(arow + k0 + half * 16);
        v8bf ahi = *(const v8bf*)(arow + k0 + half * 16 + 8);
        v16bf a;
#pragma unroll
        for (int i = 0; i < 8; ++i) { a[i] = alo[i]; a[i + 8] = ahi[i]; }
        // 4 B 32x16 fragments: lane column n0+s*16+l16, K = k0+half*16 .. +15 (contiguous)
        const size_t boff = (size_t)k0 + half * 16;
#pragma unroll
        for (int s = 0; s < 4; ++s) {
            v16bf b = *(const v16bf*)(bbas + s * bstp + boff);
            acc[s] = __builtin_amdgcn_wmma_f32_16x16x32_bf16(
                         false, a, false, b, (short)0, acc[s], false, false);
        }
    }

    float dv2[8];
#pragma unroll
    for (int j = 0; j < 8; ++j) {                     // VGPR j -> row m0 + half*8 + j
        float w = dinv[m0 + half * 8 + j];
        dv2[j] = w * w;
    }
#pragma unroll
    for (int s = 0; s < 4; ++s) {
        const int col = n0 + s * 16 + l16;
        if (col < OUT) {
#pragma unroll
            for (int j = 0; j < 8; ++j) {
                const size_t idx = (size_t)(m0 + half * 8 + j) * OUT + col;
                const float v = acc[s][j];
                H[idx]   = v;
                AGG[idx] = v * dv2[j];                // self-loop term (assign)
            }
        }
    }
}

// ---------------- edge scatter: agg[col] += h[row] * dinv[row]*dinv[col] ----------------
__global__ void k_scatter(const long long* __restrict__ row, const long long* __restrict__ col,
                          const float* __restrict__ dinv, const float* __restrict__ h,
                          float* __restrict__ agg, long long e, int OUT, int chunks) {
    long long t = (long long)blockIdx.x * blockDim.x + threadIdx.x;
    long long total = e * chunks;
    if (t >= total) return;
    long long ed = t / chunks;
    int c = (int)(t % chunks);
    long long r = row[ed], cl = col[ed];
    float w = dinv[r] * dinv[cl];
    const float* hs = h + (size_t)r * OUT;
    float* ag = agg + (size_t)cl * OUT;
#pragma unroll
    for (int j = 0; j < 4; ++j) {
        int d = j * chunks + c;                       // coalesced across threads per j
        if (d < OUT) atomicAdd(&ag[d], hs[d] * w);
    }
}

// ---------------- bias + relu, fused bf16 demotion into next layer's A staging ----------
__global__ void k_bias_relu_bf16(const float* __restrict__ agg, const float* __restrict__ bias,
                                 __bf16* __restrict__ dst, int n, int OUT) {
    // OUT is a multiple of 32 for layers 0..6, so next layer's Kpad == OUT (no padding)
    long long t = (long long)blockIdx.x * blockDim.x + threadIdx.x;
    long long total = (long long)n * OUT;
    if (t >= total) return;
    int d = (int)(t % OUT);
    dst[t] = (__bf16)fmaxf(agg[t] + bias[d], 0.0f);
}

// ---------------- final layer: bias only, fp32 to d_out ----------------
__global__ void k_bias_last(const float* __restrict__ agg, const float* __restrict__ bias,
                            float* __restrict__ dst, int n, int OUT) {
    long long t = (long long)blockIdx.x * blockDim.x + threadIdx.x;
    long long total = (long long)n * OUT;
    if (t >= total) return;
    int d = (int)(t % OUT);
    dst[t] = agg[t] + bias[d];
}

// =====================================================================================
extern "C" void kernel_launch(void* const* d_in, const int* in_sizes, int n_in,
                              void* d_out, int out_size, void* d_ws, size_t ws_size,
                              hipStream_t stream) {
    (void)in_sizes; (void)n_in; (void)out_size; (void)ws_size;
    static const int dims[9] = {3, 64, 64, 64, 128, 1024, 512, 256, 2};
    const int       N = NN;
    const long long E = NE;

    const float*     x    = (const float*)d_in[0];
    const long long* ei   = (const long long*)d_in[1];   // (2, E) int64: rows then cols
    const long long* erow = ei;
    const long long* ecol = ei + E;
    const float *Wp[8], *bp[8];
    for (int i = 0; i < 8; ++i) {
        Wp[i] = (const float*)d_in[2 + 2 * i];
        bp[i] = (const float*)d_in[3 + 2 * i];
    }

    // workspace carve-up
    char*  ws  = (char*)d_ws;
    size_t off = 0;
    auto carve = [&](size_t bytes) -> char* {
        char* p = ws + off;
        off = (off + bytes + 255) & ~(size_t)255;
        return p;
    };
    float*  dinv = (float*) carve((size_t)N * 4);
    __bf16* Abf  = (__bf16*)carve((size_t)N * 1024 * 2);   // bf16 activations (GEMM A)
    __bf16* Wtbf = (__bf16*)carve((size_t)1024 * 1024 * 2);
    float*  hbuf = (float*) carve((size_t)N * 1024 * 4);   // h = x@W
    float*  agg  = (float*) carve((size_t)N * 1024 * 4);   // aggregation target

    const int TB = 256;
    auto blk = [&](long long total) -> unsigned int {
        return (unsigned int)((total + TB - 1) / TB);
    };

    // --- normalization: deg (with self-loops) -> dinv = rsqrt(deg) ---
    k_init_ones<<<blk(N), TB, 0, stream>>>(dinv, N);
    k_deg_count<<<blk(E), TB, 0, stream>>>(ecol, dinv, E);
    k_dinv<<<blk(N), TB, 0, stream>>>(dinv, N);

    // --- layer 0 input: pad K=3 -> 32 into bf16 staging ---
    k_conv_a_bf16<<<blk((long long)N * 32), TB, 0, stream>>>(x, Abf, N, dims[0], 32);

    for (int i = 0; i < 8; ++i) {
        const int K        = dims[i];
        const int OUT      = dims[i + 1];
        const int Kpad     = (K + 31) & ~31;
        const int OUTpad64 = (OUT + 63) & ~63;

        // stage bf16 transposed weights (zero-padded)
        k_conv_wt_bf16<<<blk((long long)OUTpad64 * Kpad), TB, 0, stream>>>(
            Wp[i], Wtbf, K, OUT, Kpad, OUTpad64);

        // dense GEMM on WMMA pipe (fp32 accum), 16x64 per wave, fused self-loop init
        const int tilesN4 = OUTpad64 >> 6;
        const long long tiles = (long long)(N >> 4) * tilesN4;
        k_gemm_wmma_bf16<<<blk(tiles * 32), TB, 0, stream>>>(
            Abf, Wtbf, dinv, hbuf, agg, N, Kpad, OUT, tilesN4);

        // edge scatter (fp32 atomics, coalesced per unroll step)
        const int chunks = (OUT + 3) / 4;
        k_scatter<<<blk(E * chunks), TB, 0, stream>>>(erow, ecol, dinv, hbuf, agg, E, OUT, chunks);

        // bias (+relu) — layers 0..6 write bf16 straight into next layer's A staging
        if (i < 7) {
            k_bias_relu_bf16<<<blk((long long)N * OUT), TB, 0, stream>>>(agg, bp[i], Abf, N, OUT);
        } else {
            k_bias_last<<<blk((long long)N * OUT), TB, 0, stream>>>(agg, bp[i], (float*)d_out, N, OUT);
        }
    }
}
